// MultiHeadAttention_11613591568737
// MI455X (gfx1250) — matrix operand order repaired and checked by reading, not compiled, not run on hardware
//
#include <hip/hip_runtime.h>

// MI455X / gfx1250 multi-head attention (wide per-head projections).
// v_wmma_f32_16x16x32_f16 everywhere; B operands k-contiguous in LDS
// (ds_load_b128 frags). Tile staging: TDM tensor_load_to_lds (TENSORcnt) for
// the hot K-chunk stream, global_load_async_to_lds_b128 (ASYNCcnt) elsewhere,
// both double-buffered so DMA overlaps WMMA compute.
//
// Pipeline:
//   0) k_transpose : W{q,k,v,o} f32 [K][N] -> f16 W^T [N][K] (one-time)
//   1) k_qkv       : q/k/v projections; V stored transposed [n][t]
//   2) k_scores    : ONE pass, online softmax: P~=exp(s-m_j) f16, running
//                    max m_j -> mrun ws, rescaled rowsum l
//   3) k_pv        : vb = sum_j exp(m_{j-1}-m_j)*acc + P~_j V_j, * (1/l)
//   4) k_out       : out = cat(vb) @ Wo + bo
// Workspace (assumes ws_size >= ~853 MB).

typedef __attribute__((ext_vector_type(16))) _Float16 v16h;
typedef __attribute__((ext_vector_type(8)))  float    v8f;

constexpr int NB  = 2;
constexpr int NT  = 2048;
constexpr int NDM = 1024;
constexpr int NH  = 16;
constexpr int NHD = 64;

__device__ __forceinline__ v8f wmma_f16(v16h a, v16h b, v8f c) {
  return __builtin_amdgcn_wmma_f32_16x16x32_f16(false, a, false, b, (short)0, c,
                                                false, false);
}

// ---- CDNA5 async global->LDS copy (16B per lane), tracked by ASYNCcnt ------
__device__ __forceinline__ void async_ld_b128(void* lds_dst, const void* gbase,
                                              unsigned byte_off) {
  asm volatile("global_load_async_to_lds_b128 %0, %1, %2"
               :
               : "v"((unsigned)(size_t)lds_dst), "v"(byte_off), "s"(gbase)
               : "memory");
}
__device__ __forceinline__ void async_wait() {
  asm volatile("s_wait_asynccnt 0" ::: "memory");
}

// ---- CDNA5 Tensor Data Mover: one-instruction 2D f16 tile -> LDS -----------
#if defined(__has_builtin)
#if __has_builtin(__builtin_amdgcn_tensor_load_to_lds) && \
    __has_builtin(__builtin_amdgcn_s_wait_tensorcnt)
#define HAVE_TDM 1
#endif
#endif
#ifndef HAVE_TDM
#define HAVE_TDM 0
#endif

#if HAVE_TDM
typedef __attribute__((ext_vector_type(4))) unsigned int tdm_g0_t;
typedef __attribute__((ext_vector_type(8))) int tdm_g1_t;
typedef __attribute__((ext_vector_type(4))) int tdm_g2_t;

// Load a (64 x tile_d1) f16 tile (tile_dim0 = 64 contiguous elems per row,
// tile_d1 rows, tensor row stride stride0 elems) into LDS with each 64-half
// row padded to 72 halves: pad_interval=4 (every 32 DWORDs = 128B = one row),
// pad_amount=3 (4 DWORDs = 16B = 8 halves)  ->  matches LDK = 72 exactly.
// D# per CDNA5 ISA ch.8: g0={count,lds,global,type}, g1={dims/strides}.
// Toolchain here is the 6-arg builtin: (g0, g1, g2, g3, g4, cpol).
__device__ __forceinline__ void tdm_load_2d(const _Float16* gsrc, void* lds_dst,
                                            unsigned tile_d1, unsigned td0,
                                            unsigned td1, unsigned stride0) {
  const unsigned long long ga = (unsigned long long)(size_t)gsrc;
  tdm_g0_t g0;
  g0[0] = 1u;                                    // count = 1 (valid)
  g0[1] = (unsigned)(size_t)lds_dst;             // lds_addr (bytes)
  g0[2] = (unsigned)(ga & 0xffffffffu);          // global_addr[31:0]
  g0[3] = (unsigned)((ga >> 32) & 0x01ffffffu)   // global_addr[56:32]
          | (2u << 30);                          // type = 2 ("image")
  tdm_g1_t g1;
  g1[0] = (1 << 16)                              // data_size = 1 (2 bytes)
          | (1 << 20)                            // pad_enable
          | (4 << 22)                            // pad_interval: 32 DWORDs
          | (3 << 25);                           // pad_amount: 4 DWORDs
  g1[1] = (int)(td0 << 16);                      // tensor_dim0[15:0]
  g1[2] = (int)((td0 >> 16) | (td1 << 16));      // dim0 hi | tensor_dim1 lo
  g1[3] = (int)((td1 >> 16) | (64u << 16));      // dim1 hi | tile_dim0 = 64
  g1[4] = (int)tile_d1;                          // tile_dim1 (tile_dim2 = 0)
  g1[5] = (int)stride0;                          // tensor_dim0_stride[31:0]
  g1[6] = 0;
  g1[7] = 0;
  const tdm_g2_t gz4 = {0, 0, 0, 0};             // 2D: groups 2/3 unused
  const tdm_g1_t gz8 = {0, 0, 0, 0, 0, 0, 0, 0};
  __builtin_amdgcn_tensor_load_to_lds(g0, g1, gz4, gz4, gz8, 0);
}
#endif

// A fragment (16x32 f16): contiguous per lane -> 2x ds_load_b128.
__device__ __forceinline__ void load_a_frag(const _Float16* s, int row_base,
                                            int ld, int k0, int lane, v16h& a) {
  const int m  = row_base + (lane & 15);
  const int hi = lane >> 4;
  const _Float16* p = s + (size_t)m * ld + k0 + hi * 16;
#pragma unroll
  for (int j = 0; j < 16; ++j) a[j] = p[j];
}

// B fragment (32x16) from a tile stored [n][k] -> 2x ds_load_b128.
__device__ __forceinline__ void load_bT_frag(const _Float16* s, int n_base,
                                             int ld, int k0, int lane, v16h& b) {
  const int n  = n_base + (lane & 15);
  const int hi = lane >> 4;
  const _Float16* p = s + (size_t)n * ld + k0 + hi * 16;
#pragma unroll
  for (int j = 0; j < 16; ++j) b[j] = p[j];
}

// ---------------------------------------------------------------------------
// 0) Transpose + convert: src f32 [K][N] -> dst f16 [N][K].
// ---------------------------------------------------------------------------
__global__ __launch_bounds__(256) void k_transpose(const float* __restrict__ src,
                                                   _Float16* __restrict__ dst,
                                                   int K, int N) {
  __shared__ _Float16 Ts[64 * 72];  // [n][k]
  const int tid = threadIdx.x;
  const int n0 = blockIdx.x * 64, k0 = blockIdx.y * 64;
  for (int i = tid; i < 64 * 16; i += 256) {
    const int r = i >> 4, c = (i & 15) << 2;  // r: k, c: n
    const float4 v = *(const float4*)(src + (size_t)(k0 + r) * N + n0 + c);
    Ts[(c + 0) * 72 + r] = (_Float16)v.x;
    Ts[(c + 1) * 72 + r] = (_Float16)v.y;
    Ts[(c + 2) * 72 + r] = (_Float16)v.z;
    Ts[(c + 3) * 72 + r] = (_Float16)v.w;
  }
  __syncthreads();
  for (int i = tid; i < 64 * 8; i += 256) {
    const int r = i >> 3, c = (i & 7) << 3;  // r: n, c: k
    *(uint4*)(dst + (size_t)(n0 + r) * K + k0 + c) = *(const uint4*)(Ts + r * 72 + c);
  }
}

// ---------------------------------------------------------------------------
// 1) QKV projection: per (mat,b,h): [T,64] @ [64,1024] -> f16.
// ---------------------------------------------------------------------------
__global__ __launch_bounds__(256) void k_qkv(const float* __restrict__ x,
                                             const _Float16* __restrict__ Wqt,
                                             const _Float16* __restrict__ Wkt,
                                             const _Float16* __restrict__ Wvt,
                                             _Float16* __restrict__ qkv) {
  constexpr int LDA = 72;
  constexpr int LDB = 72;
  __shared__ _Float16 As[64 * LDA];   // x tile  [t][k]
  __shared__ _Float16 Bt[128 * LDB];  // W^T tile [n][k]; reused as C^T stage

  const int tid = threadIdx.x, lane = tid & 31, wave = tid >> 5;
  const int t0  = blockIdx.x * 64;
  const int n0  = blockIdx.y * 128;
  const int z   = blockIdx.z;
  const int mat = z >> 5;
  const int bh  = z & 31;
  const int b   = bh >> 4, h = bh & 15;

  const _Float16* Wt = (mat == 0 ? Wqt : (mat == 1 ? Wkt : Wvt));
  {
    const _Float16* base = Wt + (size_t)n0 * NDM + h * NHD;
    for (int i = tid; i < 128 * 8; i += 256) {
      const int r = i >> 3, c = (i & 7) << 3;
      async_ld_b128(Bt + r * LDB + c, base, (unsigned)((r * NDM + c) * 2));
    }
  }
  for (int i = tid; i < 64 * 16; i += 256) {
    const int r = i >> 4, c = (i & 15) << 2;
    const float4 v =
        *(const float4*)(x + (size_t)(b * NT + t0 + r) * NDM + h * NHD + c);
    As[r * LDA + c + 0] = (_Float16)v.x;
    As[r * LDA + c + 1] = (_Float16)v.y;
    As[r * LDA + c + 2] = (_Float16)v.z;
    As[r * LDA + c + 3] = (_Float16)v.w;
  }
  async_wait();
  __syncthreads();

  const int mrow = (wave & 3) << 4;
  const int nb   = (wave >> 2) << 6;
  v8f acc[4] = {};
#pragma unroll
  for (int ks = 0; ks < 2; ++ks) {
    v16h a, b0, b1, b2, b3;
    load_a_frag(As, mrow, LDA, ks * 32, lane, a);
    load_bT_frag(Bt, nb + 0, LDB, ks * 32, lane, b0);
    load_bT_frag(Bt, nb + 16, LDB, ks * 32, lane, b1);
    load_bT_frag(Bt, nb + 32, LDB, ks * 32, lane, b2);
    load_bT_frag(Bt, nb + 48, LDB, ks * 32, lane, b3);
    acc[0] = wmma_f16(a, b0, acc[0]);
    acc[1] = wmma_f16(a, b1, acc[1]);
    acc[2] = wmma_f16(a, b2, acc[2]);
    acc[3] = wmma_f16(a, b3, acc[3]);
  }
  const int col = lane & 15, hi = lane >> 4;
  if (mat < 2) {  // Q,K row-major [t][n]
    _Float16* out = qkv + (size_t)mat * NB * NH * NT * NDM + (size_t)bh * NT * NDM;
#pragma unroll
    for (int f = 0; f < 4; ++f)
#pragma unroll
      for (int i = 0; i < 8; ++i)
        out[(size_t)(t0 + mrow + i + hi * 8) * NDM + n0 + nb + f * 16 + col] =
            (_Float16)acc[f][i];
  } else {  // V transposed [n][t] via LDS restage
    _Float16* outV = qkv + (size_t)2 * NB * NH * NT * NDM + (size_t)bh * NT * NDM;
    __syncthreads();
#pragma unroll
    for (int f = 0; f < 4; ++f)
#pragma unroll
      for (int i = 0; i < 8; ++i)
        Bt[(nb + f * 16 + col) * LDB + mrow + i + hi * 8] = (_Float16)acc[f][i];
    __syncthreads();
    for (int i = tid; i < 128 * 8; i += 256) {
      const int r = i >> 3, c = (i & 7) << 3;
      *(uint4*)(outV + (size_t)(n0 + r) * NT + t0 + c) =
          *(const uint4*)(Bt + r * LDB + c);
    }
  }
}

// ---------------------------------------------------------------------------
// 2) Scores, single pass, online softmax. K chunks: TDM + double buffer.
// ---------------------------------------------------------------------------
__device__ __forceinline__ void s_tile_pipe(const _Float16* __restrict__ Kg,
                                            _Float16* Ks /*[2][64*72]*/,
                                            const _Float16* Qs, int j, int tid,
                                            int lane, int wave, int mrow,
                                            int nb, v8f& a0, v8f& a1) {
  constexpr int LDQ = NDM + 8;  // 1032
  constexpr int LDK = 72;
  const v8f zero = {};
  a0 = zero;
  a1 = zero;
#if HAVE_TDM
  if (wave == 0)  // one DMA per 64x64 chunk, padded 64->72 halves per row
    tdm_load_2d(Kg + (size_t)(j * 64) * NDM, Ks, 64, NDM, NT, NDM);
#else
  {
    const _Float16* base = Kg + (size_t)(j * 64) * NDM;
    for (int i = tid; i < 64 * 8; i += 256) {
      const int r = i >> 3, c = (i & 7) << 3;
      async_ld_b128(Ks + r * LDK + c, base, (unsigned)((r * NDM + c) * 2));
    }
  }
#endif
  for (int cc = 0; cc < 16; ++cc) {
#if HAVE_TDM
    if (wave == 0) __builtin_amdgcn_s_wait_tensorcnt(0);
#else
    async_wait();
#endif
    __syncthreads();  // chunk cc visible; buffer (cc+1)&1 free for overwrite
    if (cc + 1 < 16) {
      _Float16* dst = Ks + ((cc + 1) & 1) * (64 * LDK);
      const _Float16* base = Kg + (size_t)(j * 64) * NDM + (cc + 1) * 64;
#if HAVE_TDM
      if (wave == 0) tdm_load_2d(base, dst, 64, NDM, NT, NDM);
#else
      for (int i = tid; i < 64 * 8; i += 256) {
        const int r = i >> 3, c = (i & 7) << 3;
        async_ld_b128(dst + r * LDK + c, base, (unsigned)((r * NDM + c) * 2));
      }
#endif
    }
    const _Float16* Kt = Ks + (cc & 1) * (64 * LDK);
    const int kk = cc * 64;
    v16h aA, aB, b00, b01, b10, b11;
    load_a_frag(Qs, mrow, LDQ, kk, lane, aA);
    load_a_frag(Qs, mrow, LDQ, kk + 32, lane, aB);
    load_bT_frag(Kt, nb, LDK, 0, lane, b00);
    load_bT_frag(Kt, nb + 16, LDK, 0, lane, b01);
    load_bT_frag(Kt, nb, LDK, 32, lane, b10);
    load_bT_frag(Kt, nb + 16, LDK, 32, lane, b11);
    a0 = wmma_f16(aA, b00, a0);
    a1 = wmma_f16(aA, b01, a1);
    a0 = wmma_f16(aB, b10, a0);
    a1 = wmma_f16(aB, b11, a1);
  }
}

__global__ __launch_bounds__(256) void k_scores(const _Float16* __restrict__ Qw,
                                                const _Float16* __restrict__ Kw,
                                                _Float16* __restrict__ Pw,
                                                float* __restrict__ lw,
                                                float* __restrict__ mrun) {
  extern __shared__ __align__(16) unsigned char smem_raw[];
  constexpr int LDQ = NDM + 8;
  constexpr int LDK = 72;
  _Float16* Qs = (_Float16*)smem_raw;       // 64 x 1032
  _Float16* Ks = Qs + 64 * LDQ;             // [2][64*72]
  float* lpart  = (float*)(Ks + 2 * 64 * LDK);  // [64][2] tile max partials
  float* lspart = lpart + 128;                  // [64][2] tile sum partials
  float* mprev  = lspart + 128;                 // [64]
  float* mcur   = mprev + 64;                   // [64]
  float* lsum   = mcur + 64;                    // [64]

  const int tid = threadIdx.x, lane = tid & 31, wave = tid >> 5;
  const int t0 = blockIdx.x * 64;
  const int bh = blockIdx.y;
  const _Float16* Q = Qw + (size_t)bh * NT * NDM;
  const _Float16* K = Kw + (size_t)bh * NT * NDM;
  _Float16* P = Pw + (size_t)bh * NT * NT;

  if (tid < 64) {
    mprev[tid] = -3.0e38f;
    lsum[tid] = 0.f;
  }
  {  // resident Q tile 64x1024 (per-lane async path)
    const _Float16* base = Q + (size_t)t0 * NDM;
    for (int i = tid; i < 64 * 128; i += 256) {
      const int r = i >> 7, c = (i & 127) << 3;
      async_ld_b128(Qs + r * LDQ + c, base, (unsigned)((r * NDM + c) * 2));
    }
  }
  async_wait();
  __syncthreads();

  const int mrow = (wave & 3) << 4;
  const int nb   = (wave >> 2) << 5;
  const int col  = lane & 15, hi = lane >> 4;
  const int wcol = wave >> 2;
  const float sc = 0.125f * 1.44269504088896f;  // (1/sqrt(64)) * log2(e)

  for (int j = 0; j < NT / 64; ++j) {
    v8f a0, a1;
    s_tile_pipe(K, Ks, Qs, j, tid, lane, wave, mrow, nb, a0, a1);
#pragma unroll
    for (int i = 0; i < 8; ++i) {  // tile row-max partials
      float v = fmaxf(a0[i], a1[i]);
      v = fmaxf(v, __shfl_xor(v, 1));
      v = fmaxf(v, __shfl_xor(v, 2));
      v = fmaxf(v, __shfl_xor(v, 4));
      v = fmaxf(v, __shfl_xor(v, 8));
      if (col == 0) lpart[(mrow + i + hi * 8) * 2 + wcol] = v;
    }
    __syncthreads();
    if (tid < 64) {  // running max update + l rescale
      const float tm = fmaxf(lpart[tid * 2], lpart[tid * 2 + 1]);
      const float mold = mprev[tid];
      const float mnew = fmaxf(mold, tm);
      mprev[tid] = mnew;
      mcur[tid] = mnew;
      lsum[tid] *= exp2f((mold - mnew) * sc);
      mrun[((size_t)bh * NT + t0 + tid) * 32 + j] = mnew;
    }
    __syncthreads();
#pragma unroll
    for (int i = 0; i < 8; ++i) {
      const int row = mrow + i + hi * 8;
      const float m = mcur[row];
      const float e0 = exp2f((a0[i] - m) * sc);
      const float e1 = exp2f((a1[i] - m) * sc);
      const size_t prow = (size_t)(t0 + row) * NT + (size_t)j * 64 + nb;
      P[prow + col] = (_Float16)e0;
      P[prow + 16 + col] = (_Float16)e1;
      float s = e0 + e1;
      s += __shfl_xor(s, 1);
      s += __shfl_xor(s, 2);
      s += __shfl_xor(s, 4);
      s += __shfl_xor(s, 8);
      if (col == 0) lspart[row * 2 + wcol] = s;
    }
    __syncthreads();
    if (tid < 64) lsum[tid] += lspart[tid * 2] + lspart[tid * 2 + 1];
  }
  __syncthreads();
  if (tid < 64) lw[(size_t)bh * NT + t0 + tid] = lsum[tid];
}

// ---------------------------------------------------------------------------
// 3) vb: flash accumulation acc <- acc*exp(m_{j-1}-m_j) + P~_j V_j, * 1/l.
// ---------------------------------------------------------------------------
__global__ __launch_bounds__(256) void k_pv(const _Float16* __restrict__ Pw,
                                            const _Float16* __restrict__ Vtw,
                                            const float* __restrict__ lw,
                                            const float* __restrict__ mrun,
                                            _Float16* __restrict__ vbw) {
  constexpr int LDP = 72;
  constexpr int LDV = 72;
  __shared__ _Float16 Ps[2][64 * LDP];
  __shared__ _Float16 Vs[2][128 * LDV];
  __shared__ float linv[64];
  __shared__ float Ms[64 * 32];

  const int tid = threadIdx.x, lane = tid & 31, wave = tid >> 5;
  const int t0 = blockIdx.x * 64;
  const int n0 = blockIdx.y * 128;
  const int bh = blockIdx.z;
  const _Float16* Pb = Pw + (size_t)bh * NT * NT;
  const _Float16* Vb = Vtw + (size_t)bh * NT * NDM;  // [n][t]
  _Float16* outb = vbw + (size_t)bh * NT * NDM;

  const int mrow = (wave & 3) << 4;
  const int nb   = (wave >> 2) << 6;
  const int col = lane & 15, hi = lane >> 4;
  const float sc = 0.125f * 1.44269504088896f;

  auto issue = [&](int j, int buf) {
    const _Float16* baseP = Pb + (size_t)t0 * NT + (size_t)j * 64;
    for (int i = tid; i < 64 * 8; i += 256) {
      const int r = i >> 3, c = (i & 7) << 3;
      async_ld_b128(&Ps[buf][r * LDP + c], baseP, (unsigned)((r * NT + c) * 2));
    }
    const _Float16* baseV = Vb + (size_t)n0 * NT + (size_t)j * 64;
    for (int i = tid; i < 128 * 8; i += 256) {
      const int r = i >> 3, c = (i & 7) << 3;
      async_ld_b128(&Vs[buf][r * LDV + c], baseV, (unsigned)((r * NT + c) * 2));
    }
  };

  if (tid < 64) linv[tid] = 1.0f / lw[(size_t)bh * NT + t0 + tid];
  for (int i = tid; i < 64 * 32; i += 256)
    Ms[i] = mrun[((size_t)bh * NT + t0 + (i >> 5)) * 32 + (i & 31)];
  issue(0, 0);

  v8f acc[4] = {};
  for (int j = 0; j < NT / 64; ++j) {
    async_wait();
    __syncthreads();
    if (j + 1 < NT / 64) issue(j + 1, (j + 1) & 1);
    if (j > 0) {  // flash rescale of accumulators
#pragma unroll
      for (int i = 0; i < 8; ++i) {
        const int row = mrow + i + hi * 8;
        const float r = exp2f((Ms[row * 32 + j - 1] - Ms[row * 32 + j]) * sc);
        acc[0][i] *= r;
        acc[1][i] *= r;
        acc[2][i] *= r;
        acc[3][i] *= r;
      }
    }
    const _Float16* Pt = Ps[j & 1];
    const _Float16* Vt = Vs[j & 1];
#pragma unroll
    for (int ks = 0; ks < 2; ++ks) {
      v16h a, b0, b1, b2, b3;
      load_a_frag(Pt, mrow, LDP, ks * 32, lane, a);
      load_bT_frag(Vt, nb + 0, LDV, ks * 32, lane, b0);
      load_bT_frag(Vt, nb + 16, LDV, ks * 32, lane, b1);
      load_bT_frag(Vt, nb + 32, LDV, ks * 32, lane, b2);
      load_bT_frag(Vt, nb + 48, LDV, ks * 32, lane, b3);
      acc[0] = wmma_f16(a, b0, acc[0]);
      acc[1] = wmma_f16(a, b1, acc[1]);
      acc[2] = wmma_f16(a, b2, acc[2]);
      acc[3] = wmma_f16(a, b3, acc[3]);
    }
  }
#pragma unroll
  for (int f = 0; f < 4; ++f)
#pragma unroll
    for (int i = 0; i < 8; ++i) {
      const int row = mrow + i + hi * 8;
      outb[(size_t)(t0 + row) * NDM + n0 + nb + f * 16 + col] =
          (_Float16)(acc[f][i] * linv[row]);
    }
}

// ---------------------------------------------------------------------------
// 4) out = cat(vb) @ Wo + bo; tiles double-buffered + async.
// ---------------------------------------------------------------------------
__global__ __launch_bounds__(256) void k_out(const _Float16* __restrict__ vbw,
                                             const _Float16* __restrict__ Wot,
                                             const float* __restrict__ bo,
                                             float* __restrict__ out) {
  constexpr int LDA = 72;
  constexpr int LDB = 72;
  __shared__ _Float16 As[2][64 * LDA];   // [t][k]
  __shared__ _Float16 Bt[2][128 * LDB];  // [n][k]

  const int tid = threadIdx.x, lane = tid & 31, wave = tid >> 5;
  const int r0 = blockIdx.x * 64;
  const int n0 = blockIdx.y * 128;
  const int b  = r0 >> 11;
  const int t0 = r0 & 2047;

  const int mrow = (wave & 3) << 4;
  const int nb   = (wave >> 2) << 6;

  auto issue = [&](int kt, int buf) {
    const int kk = kt * 64;
    const int h = kk >> 10, m0 = kk & 1023;
    const _Float16* baseA = vbw + ((size_t)(b * NH + h) * NT + t0) * NDM + m0;
    for (int i = tid; i < 64 * 8; i += 256) {
      const int r = i >> 3, c = (i & 7) << 3;
      async_ld_b128(&As[buf][r * LDA + c], baseA, (unsigned)((r * NDM + c) * 2));
    }
    const _Float16* baseB = Wot + (size_t)n0 * (NH * NDM) + kk;
    for (int i = tid; i < 128 * 8; i += 256) {
      const int r = i >> 3, c = (i & 7) << 3;
      async_ld_b128(&Bt[buf][r * LDB + c], baseB,
                    (unsigned)((r * NH * NDM + c) * 2));
    }
  };

  issue(0, 0);
  v8f acc[4] = {};
  for (int kt = 0; kt < (NH * NDM) / 64; ++kt) {
    async_wait();
    __syncthreads();
    if (kt + 1 < (NH * NDM) / 64) issue(kt + 1, (kt + 1) & 1);
    const _Float16* At = As[kt & 1];
    const _Float16* Bb = Bt[kt & 1];
#pragma unroll
    for (int ks = 0; ks < 2; ++ks) {
      v16h a, b0, b1, b2, b3;
      load_a_frag(At, mrow, LDA, ks * 32, lane, a);
      load_bT_frag(Bb, nb + 0, LDB, ks * 32, lane, b0);
      load_bT_frag(Bb, nb + 16, LDB, ks * 32, lane, b1);
      load_bT_frag(Bb, nb + 32, LDB, ks * 32, lane, b2);
      load_bT_frag(Bb, nb + 48, LDB, ks * 32, lane, b3);
      acc[0] = wmma_f16(a, b0, acc[0]);
      acc[1] = wmma_f16(a, b1, acc[1]);
      acc[2] = wmma_f16(a, b2, acc[2]);
      acc[3] = wmma_f16(a, b3, acc[3]);
    }
  }
  const int col = lane & 15, hi = lane >> 4;
#pragma unroll
  for (int f = 0; f < 4; ++f)
#pragma unroll
    for (int i = 0; i < 8; ++i) {
      const int n = n0 + nb + f * 16 + col;
      out[(size_t)(r0 + mrow + i + hi * 8) * NDM + n] = acc[f][i] + bo[n];
    }
}

// ---------------------------------------------------------------------------
extern "C" void kernel_launch(void* const* d_in, const int* in_sizes, int n_in,
                              void* d_out, int out_size, void* d_ws,
                              size_t ws_size, hipStream_t stream) {
  (void)in_sizes; (void)n_in; (void)out_size; (void)ws_size;
  const float* x  = (const float*)d_in[0];
  const float* Wq = (const float*)d_in[1];
  const float* Wk = (const float*)d_in[2];
  const float* Wv = (const float*)d_in[3];
  const float* Wo = (const float*)d_in[4];
  const float* bo = (const float*)d_in[5];
  float* out = (float*)d_out;

  const size_t QKV_ELEMS = (size_t)NB * NH * NT * NDM;  // 67,108,864
  _Float16* Qw  = (_Float16*)d_ws;
  _Float16* Kw  = Qw + QKV_ELEMS;
  _Float16* Vtw = Kw + QKV_ELEMS;                 // [bh][n][t]
  _Float16* Pw  = Vtw + QKV_ELEMS;                // [bh][t][key]
  _Float16* VBw = Pw + (size_t)NB * NH * NT * NT;
  float*    lw  = (float*)(VBw + QKV_ELEMS);
  _Float16* Wqt = (_Float16*)(lw + (size_t)NB * NH * NT);  // [1024][1024]
  _Float16* Wkt = Wqt + (size_t)NDM * NDM;
  _Float16* Wvt = Wkt + (size_t)NDM * NDM;
  _Float16* Wot = Wvt + (size_t)NDM * NDM;                 // [1024][16384]
  float*    mrw = (float*)(Wot + (size_t)NDM * NH * NDM);  // [bh*T][32]

  k_transpose<<<dim3(16, 16), 256, 0, stream>>>(Wq, Wqt, NDM, NDM);
  k_transpose<<<dim3(16, 16), 256, 0, stream>>>(Wk, Wkt, NDM, NDM);
  k_transpose<<<dim3(16, 16), 256, 0, stream>>>(Wv, Wvt, NDM, NDM);
  k_transpose<<<dim3(16, 256), 256, 0, stream>>>(Wo, Wot, NH * NDM, NDM);

  k_qkv<<<dim3(NT / 64, NDM / 128, 3 * NB * NH), 256, 0, stream>>>(x, Wqt, Wkt,
                                                                   Wvt, Qw);

  const size_t shB = (size_t)(64 * (NDM + 8) + 2 * 64 * 72) * sizeof(_Float16) +
                     (size_t)(128 + 128 + 64 + 64 + 64) * sizeof(float);
  k_scores<<<dim3(NT / 64, NB * NH), 256, shB, stream>>>(Qw, Kw, Pw, lw, mrw);

  k_pv<<<dim3(NT / 64, NDM / 128, NB * NH), 256, 0, stream>>>(Pw, Vtw, lw, mrw,
                                                              VBw);

  k_out<<<dim3(NB * NT / 64, NDM / 128), 256, 0, stream>>>(VBw, Wot, bo, out);
}
